// GATConv_57999238365771
// MI455X (gfx1250) — compile-verified, matrix operand order checked by reading, not run on hardware
//
#include <hip/hip_runtime.h>

#define LRELU_ALPHA 0.2f
#define NN   8192
#define FIN  512
#define FOUT 256

typedef __attribute__((ext_vector_type(16))) __bf16          v16bf;
typedef __attribute__((ext_vector_type(8)))  unsigned short  u16x8;
typedef __attribute__((ext_vector_type(8)))  float           v8f;

union ABu {
    v16bf b;     // WMMA operand view
    u16x8 h[2];  // two 16B halves for b128 loads
};

// ---------------------------------------------------------------------------
// Kernel 1: h = x @ W  (bf16 WMMA, f32 accum). Also writes hT (bf16, K-major)
// for fast B-fragment loads in the attention matmul.
// One 16x16 output tile per wave; 4 waves per block.
// ---------------------------------------------------------------------------
__global__ __launch_bounds__(128) void gemm_h_kernel(
        const float* __restrict__ x, const float* __restrict__ W,
        float* __restrict__ h, unsigned short* __restrict__ hT) {
    const int lane = threadIdx.x & 31;
    const int wid  = threadIdx.x >> 5;
    const int tile = blockIdx.x * 4 + wid;          // 8192 tiles total
    const int mtile = tile >> 4, ntile = tile & 15;
    const int i0 = mtile * 16, n0 = ntile * 16;
    const int half = lane >> 4;                     // K sub-tile select
    const int lr   = lane & 15;
    const int i = i0 + lr;                          // A row for this lane
    const int n = n0 + lr;                          // B col for this lane

    v8f c = {};
    for (int ks = 0; ks < FIN / 32; ++ks) {
        const int k0 = ks * 32;
        const int ka = k0 + (half ? 8 : 0);         // lanes 16-31 take K+8
        ABu A, B;
        const float* xr = &x[(size_t)i * FIN + ka];
#pragma unroll
        for (int e = 0; e < 8; ++e) {
            A.b[e]     = (__bf16)xr[e];             // K = ka+e
            A.b[e + 8] = (__bf16)xr[16 + e];        // K = ka+16+e
            B.b[e]     = (__bf16)W[(size_t)(ka + e)      * FOUT + n];
            B.b[e + 8] = (__bf16)W[(size_t)(ka + 16 + e) * FOUT + n];
        }
        c = __builtin_amdgcn_wmma_f32_16x16x32_bf16(
                false, A.b, false, B.b, (short)0, c, false, false);
    }
#pragma unroll
    for (int r = 0; r < 8; ++r) {
        const int row = i0 + r + (half ? 8 : 0);
        const int col = n0 + lr;
        const float v = c[r];
        h[(size_t)row * FOUT + col] = v;
        __bf16 vb = (__bf16)v;
        hT[(size_t)col * NN + row]  = *(unsigned short*)&vb;
    }
}

// ---------------------------------------------------------------------------
// Kernel 2: f1[i] = h[i,:] . a[0:256], f2[i] = h[i,:] . a[256:512]
// One row per wave (wave32 shuffle reduction).
// ---------------------------------------------------------------------------
__global__ __launch_bounds__(256) void f12_kernel(
        const float* __restrict__ h, const float* __restrict__ a,
        float* __restrict__ f1, float* __restrict__ f2) {
    const int lane = threadIdx.x & 31;
    const int wid  = threadIdx.x >> 5;
    const int i = blockIdx.x * 8 + wid;
    float s1 = 0.f, s2 = 0.f;
    for (int cidx = lane; cidx < FOUT; cidx += 32) {
        const float hv = h[(size_t)i * FOUT + cidx];
        s1 += hv * a[cidx];
        s2 += hv * a[FOUT + cidx];
    }
#pragma unroll
    for (int off = 16; off > 0; off >>= 1) {
        s1 += __shfl_xor(s1, off, 32);
        s2 += __shfl_xor(s2, off, 32);
    }
    if (lane == 0) { f1[i] = s1; f2[i] = s2; }
}

// ---------------------------------------------------------------------------
// Kernel 3: per-row softmax stats (online max + rescaled sum) in one pass
// over the 256MB adj matrix. Also bit-packs adj (wave32 ballot) into bm
// (8 MB, L2-resident) so the WMMA pass never touches adj again.
// One row per block; wave w covers j = k*256 + w*32 + lane (contiguous 32).
// ---------------------------------------------------------------------------
__global__ __launch_bounds__(256) void stats_kernel(
        const int* __restrict__ adj, const float* __restrict__ f1,
        const float* __restrict__ f2,
        float* __restrict__ mrow, float* __restrict__ invZ,
        unsigned int* __restrict__ bm) {
    const int i    = blockIdx.x;
    const int t    = threadIdx.x;
    const int lane = t & 31;
    const int w    = t >> 5;
    const float f1i = f1[i];
    float m = -3.4e38f, s = 0.f;
    for (int k = 0; k < NN / 256; ++k) {
        const int j  = t + k * 256;
        const int av = adj[(size_t)i * NN + j];
        // bit-pack: wave w's 32 lanes cover dword j/32 == k*8 + w
        const unsigned mask = __builtin_amdgcn_ballot_w32(av > 0);
        if (lane == 0) bm[(size_t)i * (NN / 32) + k * 8 + w] = mask;
        float e = f1i + f2[j];
        e = (e >= 0.f) ? e : LRELU_ALPHA * e;
        const float val = (av > 0) ? e : 0.f;       // masked -> 0.0 (reference bug kept)
        if (val > m) { s = s * __expf(m - val) + 1.f; m = val; }
        else         { s += __expf(val - m); }
    }
    __shared__ float sm[256], ss[256];
    sm[t] = m; ss[t] = s;
    __syncthreads();
    for (int st = 128; st > 0; st >>= 1) {
        if (t < st) {
            const float m2 = sm[t + st], s2 = ss[t + st];
            const float mm = fmaxf(sm[t], m2);
            ss[t] = ss[t] * __expf(sm[t] - mm) + s2 * __expf(m2 - mm);
            sm[t] = mm;
        }
        __syncthreads();
    }
    if (t == 0) { mrow[i] = sm[0]; invZ[i] = 1.f / ss[0]; }
}

// ---------------------------------------------------------------------------
// Kernel 4: out = elu(softmax_row(att) @ h). Attention probs regenerated on
// the fly from the bitmask + f1/f2/stats, staged as a double-buffered bf16
// LDS tile, consumed by v_wmma_f32_16x16x32_bf16 (one barrier per K-step).
// Block = 16 output rows x 256 cols; 8 waves, 2 N-tiles each.
// ---------------------------------------------------------------------------
__global__ __launch_bounds__(256) void atth_kernel(
        const unsigned int* __restrict__ bm, const float* __restrict__ f1,
        const float* __restrict__ f2, const float* __restrict__ mrow,
        const float* __restrict__ invZ, const unsigned short* __restrict__ hT,
        float* __restrict__ out) {
    __shared__ unsigned short P[2][16][32];         // double-buffered bf16 probs
    const int t    = threadIdx.x;
    const int lane = t & 31;
    const int wid  = t >> 5;
    const int half = lane >> 4;
    const int lr   = lane & 15;
    const int i0   = blockIdx.x * 16;

    // P-tile production: wave pm owns rows pm and pm+8; lane -> column pk
    const int pm = t >> 5;                          // 0..7
    const int pk = t & 31;
    const int ra = i0 + pm, rb = i0 + pm + 8;
    const float f1a = f1[ra], ma = mrow[ra], iza = invZ[ra];
    const float f1b = f1[rb], mb = mrow[rb], izb = invZ[rb];
    const unsigned int* bma_row = &bm[(size_t)ra * (NN / 32)];
    const unsigned int* bmb_row = &bm[(size_t)rb * (NN / 32)];

    const int n0a = (wid * 2) * 16;
    const int n0b = (wid * 2 + 1) * 16;
    const size_t bbA = (size_t)(n0a + lr) * NN;     // hT column base (K-major)
    const size_t bbB = (size_t)(n0b + lr) * NN;
    const int kAoff = half ? 8 : 0;                 // 16-bit A/B lane-half K offset

    v8f c0 = {}, c1 = {};
    for (int kt = 0; kt < NN / 32; ++kt) {
        const int buf = kt & 1;
        const int k0  = kt * 32;
        const int j   = k0 + pk;
        const float f2j = f2[j];
        const unsigned bva = bma_row[kt];           // wave-uniform dword
        const unsigned bvb = bmb_row[kt];
        float ea = f1a + f2j; ea = (ea >= 0.f) ? ea : LRELU_ALPHA * ea;
        float eb = f1b + f2j; eb = (eb >= 0.f) ? eb : LRELU_ALPHA * eb;
        const float va = ((bva >> pk) & 1u) ? ea : 0.f;
        const float vb = ((bvb >> pk) & 1u) ? eb : 0.f;
        __bf16 pa = (__bf16)(__expf(va - ma) * iza);
        __bf16 pb = (__bf16)(__expf(vb - mb) * izb);
        P[buf][pm][pk]     = *(unsigned short*)&pa;
        P[buf][pm + 8][pk] = *(unsigned short*)&pb;
        __syncthreads();                            // single barrier per iter
                                                    // (double buffer protects WAR)
        ABu A, B0, B1;
        A.h[0]  = *(const u16x8*)&P[buf][lr][kAoff];          // ds_load_b128
        A.h[1]  = *(const u16x8*)&P[buf][lr][kAoff + 16];
        B0.h[0] = *(const u16x8*)&hT[bbA + k0 + kAoff];       // global b128
        B0.h[1] = *(const u16x8*)&hT[bbA + k0 + kAoff + 16];
        B1.h[0] = *(const u16x8*)&hT[bbB + k0 + kAoff];
        B1.h[1] = *(const u16x8*)&hT[bbB + k0 + kAoff + 16];
        c0 = __builtin_amdgcn_wmma_f32_16x16x32_bf16(
                 false, A.b, false, B0.b, (short)0, c0, false, false);
        c1 = __builtin_amdgcn_wmma_f32_16x16x32_bf16(
                 false, A.b, false, B1.b, (short)0, c1, false, false);
    }
#pragma unroll
    for (int r = 0; r < 8; ++r) {
        const int row = i0 + r + (half ? 8 : 0);
        float v0 = c0[r]; v0 = (v0 > 0.f) ? v0 : (__expf(v0) - 1.f);   // ELU
        float v1 = c1[r]; v1 = (v1 > 0.f) ? v1 : (__expf(v1) - 1.f);
        out[(size_t)row * FOUT + n0a + lr] = v0;
        out[(size_t)row * FOUT + n0b + lr] = v1;
    }
}

// ---------------------------------------------------------------------------
extern "C" void kernel_launch(void* const* d_in, const int* in_sizes, int n_in,
                              void* d_out, int out_size, void* d_ws, size_t ws_size,
                              hipStream_t stream) {
    const float* x   = (const float*)d_in[0];   // [8192,512]
    const int*   adj = (const int*)  d_in[1];   // [8192,8192]
    const float* W   = (const float*)d_in[2];   // [512,256]
    const float* a   = (const float*)d_in[3];   // [512,1]
    float* out = (float*)d_out;                 // [8192,256]

    char* ws = (char*)d_ws;
    float*          h    = (float*)ws;                                        // 8 MB
    unsigned short* hT   = (unsigned short*)(ws + (size_t)8  * 1024 * 1024);  // 4 MB
    unsigned int*   bm   = (unsigned int*)  (ws + (size_t)12 * 1024 * 1024);  // 8 MB
    float*          f1   = (float*)(ws + (size_t)20 * 1024 * 1024);
    float*          f2   = f1 + NN;
    float*          mrow = f2 + NN;
    float*          invZ = mrow + NN;

    gemm_h_kernel<<<2048, 128, 0, stream>>>(x, W, h, hT);
    f12_kernel   <<<1024, 256, 0, stream>>>(h, a, f1, f2);
    stats_kernel <<<NN,   256, 0, stream>>>(adj, f1, f2, mrow, invZ, bm);
    atth_kernel  <<<512,  256, 0, stream>>>(bm, f1, f2, mrow, invZ, hT, out);
}
